// UPSA_22789096473161
// MI455X (gfx1250) — matrix operand order repaired and checked by reading, not compiled, not run on hardware
//
#include <hip/hip_runtime.h>
#include <hip/hip_bf16.h>

// Problem constants (match reference)
#define BB 16
#define PP 16
#define NN 4096
#define CC 768
#define HH 12
#define DD 64

typedef __attribute__((ext_vector_type(8)))  __bf16 v8bf;
typedef __attribute__((ext_vector_type(16))) __bf16 v16bf;
typedef __attribute__((ext_vector_type(8)))  float  v8f;

static __device__ inline v8f vzero8() {
  v8f r;
#pragma unroll
  for (int i = 0; i < 8; ++i) r[i] = 0.0f;
  return r;
}

static __device__ inline v16bf cat16(v8bf lo, v8bf hi) {
  return __builtin_shufflevector(lo, hi, 0, 1, 2, 3, 4, 5, 6, 7, 8, 9, 10, 11, 12, 13, 14, 15);
}

// 8 contiguous fp32 -> v8bf
static __device__ inline v8bf cvt8(const float* p) {
  float4 a = *(const float4*)p;
  float4 b = *(const float4*)(p + 4);
  v8bf r;
  r[0] = (__bf16)a.x; r[1] = (__bf16)a.y; r[2] = (__bf16)a.z; r[3] = (__bf16)a.w;
  r[4] = (__bf16)b.x; r[5] = (__bf16)b.y; r[6] = (__bf16)b.z; r[7] = (__bf16)b.w;
  return r;
}

// A operand (16x32 MxK). Per CDNA5 ISA layout: lane holds row M=lane%16;
// K chunks {0..7,16..23} for lanes 0-15, {8..15,24..31} for lanes 16-31.
static __device__ inline v16bf load_a_f32(const float* rowk0, unsigned lane) {
  const float* p = rowk0 + ((lane & 16u) ? 8 : 0);
  return cat16(cvt8(p), cvt8(p + 16));
}

// B operand (32x16 KxN), source is column-major (32 contiguous K per column).
// Lane holds column N=lane%16; K chunks {0..7,16..23} (lanes 0-15) or {8..15,24..31} (lanes 16-31).
static __device__ inline v16bf load_b_bf16(const __bf16* colk0, unsigned lane) {
  const __bf16* p = colk0 + ((lane & 16u) ? 8 : 0);
  return cat16(*(const v8bf*)p, *(const v8bf*)(p + 16));
}

static __device__ inline v8f wmma_bf16(v16bf a, v16bf b, v8f c) {
  return __builtin_amdgcn_wmma_f32_16x16x32_bf16(false, a, false, b, (short)0, c, false, false);
}

// CDNA5 async global->LDS copy (16B), tracked by ASYNCcnt (CDNA5 ISA)
static __device__ inline void async_copy_b128(const __bf16* gsrc, __bf16* lds_dst) {
  unsigned ldsoff = (unsigned)(unsigned long long)lds_dst;  // low 32b of generic = LDS offset
  asm volatile("global_load_async_to_lds_b128 %0, %1, off"
               :: "v"(ldsoff), "v"(gsrc)
               : "memory");
}
static __device__ inline void wait_asynccnt0() {
  asm volatile("s_wait_asynccnt 0x0" ::: "memory");
}

// ---------------------------------------------------------------------------
// Kernel 0: fp32 -> bf16 weight pre-convert (kv_w, proj_w), done once.
// ---------------------------------------------------------------------------
__global__ __launch_bounds__(256) void cvt_f32_bf16_kernel(
    const float* __restrict__ src, __bf16* __restrict__ dst, int n8) {
  int i = blockIdx.x * 256 + threadIdx.x;
  if (i < n8) *(v8bf*)(dst + (size_t)i * 8) = cvt8(src + (size_t)i * 8);
}

// ---------------------------------------------------------------------------
// Kernel 1: kv = x @ kv_w.T    (M=B*N=65536, K=768, Nout=1536)
// Block: 256 threads = 8 waves, tile 128(M) x 128(N).  kv_w tile (bf16,
// pre-converted) streamed into double-buffered LDS with async global->LDS
// copies overlapping the WMMA work.  N-tile of 128 == 2 full heads, so the
// k half is L2-normalized in-register and stored [B,H,N,D]; v is stored
// transposed [B,H,D,N] with packed 16B stores.
// ---------------------------------------------------------------------------
__global__ __launch_bounds__(256) void kv_gemm_kernel(
    const float* __restrict__ x, const __bf16* __restrict__ kvw_bf,
    __bf16* __restrict__ k_out, __bf16* __restrict__ v_out) {
  __shared__ __bf16 Bs[2][128 * 32];  // [buf][col][k] col-major, 8 KB each

  const unsigned tid  = threadIdx.x;
  const unsigned lane = tid & 31u;
  const unsigned wave = tid >> 5;
  const unsigned nb   = blockIdx.x % 12u;
  const unsigned mb   = blockIdx.x / 12u;
  const unsigned n0   = nb * 128u;              // column into kv [0,1536)
  const size_t   m0   = (size_t)mb * 128u;      // flat row into [B*N]
  const unsigned b    = (unsigned)(m0 / NN);
  const unsigned tok0 = (unsigned)(m0 % NN);

  v8f acc[8];
#pragma unroll
  for (int t = 0; t < 8; ++t) acc[t] = vzero8();

  // async B-tile fill: 512 x 16B chunks per K-step, 2 per thread
  // chunk ch: col = ch>>2, part = ch&3 (8 halves each)
  const unsigned ch0 = tid, ch1 = tid + 256u;
  const __bf16* wbase = kvw_bf + (size_t)n0 * CC;

  auto issueB = [&](int k0, int buf) {
#pragma unroll
    for (int j = 0; j < 2; ++j) {
      unsigned ch = j ? ch1 : ch0;
      unsigned col = ch >> 2, part = ch & 3u;
      async_copy_b128(wbase + (size_t)col * CC + k0 + part * 8u,
                      &Bs[buf][col * 32u + part * 8u]);
    }
  };

  const float* abase = x + (m0 + wave * 16u + (lane & 15u)) * CC;

  issueB(0, 0);
  wait_asynccnt0();
  __syncthreads();

  for (int k0 = 0; k0 < CC; k0 += 32) {
    const int buf = (k0 >> 5) & 1;
    if (k0 + 32 < CC) issueB(k0 + 32, buf ^ 1);           // prefetch next tile
    if (k0 + 64 < CC) __builtin_prefetch(abase + k0 + 64, 0, 1);

    v16bf a = load_a_f32(abase + k0, lane);
    v16bf bm[8];
#pragma unroll
    for (int t = 0; t < 8; ++t)
      bm[t] = load_b_bf16(&Bs[buf][(t * 16u + (lane & 15u)) * 32u], lane);
#pragma unroll
    for (int t = 0; t < 8; ++t) acc[t] = wmma_bf16(a, bm[t], acc[t]);

    wait_asynccnt0();   // our async copies into buf^1 landed
    __syncthreads();    // everyone's did; buf may be overwritten next iter
  }

  const unsigned rowoff = (lane & 16u) ? 8u : 0u;  // D-layout: M = r + rowoff
  const unsigned c = lane & 15u;                   // D-layout: N = lane%16

  if (n0 < CC) {
    // k heads: normalize each row over D=64 (4 subtiles per head)
    const unsigned h0 = n0 / DD;
#pragma unroll
    for (int g = 0; g < 2; ++g) {
#pragma unroll
      for (int r = 0; r < 8; ++r) {
        float ss = 0.0f;
#pragma unroll
        for (int t = 0; t < 4; ++t) { float v = acc[g * 4 + t][r]; ss += v * v; }
        ss += __shfl_xor(ss, 1);
        ss += __shfl_xor(ss, 2);
        ss += __shfl_xor(ss, 4);
        ss += __shfl_xor(ss, 8);
        float inv = 1.0f / fmaxf(sqrtf(ss), 1e-12f);
        unsigned tok = tok0 + wave * 16u + rowoff + (unsigned)r;
        __bf16* dst = k_out + ((size_t)(b * HH + h0 + g) * NN + tok) * DD + c;
#pragma unroll
        for (int t = 0; t < 4; ++t)
          dst[t * 16] = (__bf16)(acc[g * 4 + t][r] * inv);
      }
    }
  } else {
    // v heads: transpose to [B,H,D,N]; 8 consecutive tokens pack per lane
    const unsigned h0 = (n0 - CC) / DD;
#pragma unroll
    for (int g = 0; g < 2; ++g) {
#pragma unroll
      for (int t = 0; t < 4; ++t) {
        v8bf pk;
#pragma unroll
        for (int r = 0; r < 8; ++r) pk[r] = (__bf16)acc[g * 4 + t][r];
        unsigned d = (unsigned)t * 16u + c;
        size_t base = ((size_t)(b * HH + h0 + g) * DD + d) * NN + tok0 + wave * 16u + rowoff;
        *(v8bf*)(v_out + base) = pk;
      }
    }
  }
}

// ---------------------------------------------------------------------------
// Kernel 2: flash attention per (b,h).  One wave, P=16, online softmax over
// N=4096 in 32-token steps.  q normalized in-register; logits 2xWMMA,
// P restaged through LDS into A-operand swizzle, P@V 4xWMMA.
// ---------------------------------------------------------------------------
__global__ __launch_bounds__(32) void attn_kernel(
    const float* __restrict__ lpn_q, const __bf16* __restrict__ k_bf,
    const __bf16* __restrict__ v_t, const float* __restrict__ logit_scale,
    float* __restrict__ attn_out) {
  __shared__ __bf16 Ps[16 * 32];  // [row p][col n] row-major

  const unsigned lane = threadIdx.x & 31u;
  const unsigned bh = blockIdx.x;
  const unsigned b = bh / HH, h = bh % HH;
  const unsigned rowoff = (lane & 16u) ? 8u : 0u;
  const unsigned c = lane & 15u;

  // load q row (this lane's A-operand slice), L2-normalize over D=64
  const float* qbase = lpn_q + ((size_t)(b * PP) + c) * CC + h * DD;
  float qv[32];
#pragma unroll
  for (int kc = 0; kc < 2; ++kc) {
    const float* p = qbase + kc * 32 + ((lane & 16u) ? 8 : 0);
#pragma unroll
    for (int i = 0; i < 8; ++i) {
      qv[kc * 16 + i]     = p[i];
      qv[kc * 16 + 8 + i] = p[16 + i];
    }
  }
  float ss = 0.0f;
#pragma unroll
  for (int i = 0; i < 32; ++i) ss += qv[i] * qv[i];
  ss += __shfl_xor(ss, 16);  // partner lane holds the other half of the row
  float qinv = 1.0f / fmaxf(sqrtf(ss), 1e-12f);
  v16bf aq[2];
#pragma unroll
  for (int kc = 0; kc < 2; ++kc)
#pragma unroll
    for (int i = 0; i < 16; ++i) aq[kc][i] = (__bf16)(qv[kc * 16 + i] * qinv);

  const float sc = __expf(fminf(logit_scale[h], 4.60517018598809136804f));  // ln(100)

  float M[8], L[8];
#pragma unroll
  for (int r = 0; r < 8; ++r) { M[r] = -1e30f; L[r] = 0.0f; }
  v8f O[4];
#pragma unroll
  for (int t = 0; t < 4; ++t) O[t] = vzero8();

  const __bf16* kb = k_bf + (size_t)bh * NN * DD;
  const __bf16* vb = v_t + (size_t)bh * DD * NN;

  for (int n0 = 0; n0 < NN; n0 += 32) {
    v8f S[2];
    S[0] = vzero8();
    S[1] = vzero8();
#pragma unroll
    for (int tt = 0; tt < 2; ++tt) {
      const __bf16* kcol = kb + (size_t)((unsigned)n0 + (unsigned)tt * 16u + c) * DD;
#pragma unroll
      for (int dc = 0; dc < 2; ++dc) {
        v16bf bm = load_b_bf16(kcol + dc * 32, lane);
        S[tt] = wmma_bf16(aq[dc], bm, S[tt]);
      }
    }
    float al[8];
#pragma unroll
    for (int r = 0; r < 8; ++r) {
      float s0 = S[0][r] * sc, s1 = S[1][r] * sc;
      float mx = fmaxf(s0, s1);
      mx = fmaxf(mx, __shfl_xor(mx, 1));
      mx = fmaxf(mx, __shfl_xor(mx, 2));
      mx = fmaxf(mx, __shfl_xor(mx, 4));
      mx = fmaxf(mx, __shfl_xor(mx, 8));
      float Mn = fmaxf(M[r], mx);
      float a = __expf(M[r] - Mn);
      float p0 = __expf(s0 - Mn), p1 = __expf(s1 - Mn);
      float rs = p0 + p1;
      rs += __shfl_xor(rs, 1);
      rs += __shfl_xor(rs, 2);
      rs += __shfl_xor(rs, 4);
      rs += __shfl_xor(rs, 8);
      L[r] = L[r] * a + rs;
      M[r] = Mn;
      al[r] = a;
      Ps[(rowoff + r) * 32u + c]       = (__bf16)p0;
      Ps[(rowoff + r) * 32u + 16u + c] = (__bf16)p1;
    }
#pragma unroll
    for (int t = 0; t < 4; ++t)
#pragma unroll
      for (int r = 0; r < 8; ++r) O[t][r] *= al[r];
    __syncthreads();
    // P tile as A operand from LDS (A-layout K chunks {0..7,16..23})
    const __bf16* prow = Ps + c * 32u + ((lane & 16u) ? 8 : 0);
    v16bf ap = cat16(*(const v8bf*)prow, *(const v8bf*)(prow + 16));
#pragma unroll
    for (int t = 0; t < 4; ++t) {
      const __bf16* vcol = vb + (size_t)((unsigned)t * 16u + c) * NN + n0;
      v16bf bm = load_b_bf16(vcol, lane);
      O[t] = wmma_bf16(ap, bm, O[t]);
    }
    __syncthreads();
  }

#pragma unroll
  for (int r = 0; r < 8; ++r) {
    float invl = 1.0f / L[r];
    size_t row = ((size_t)(b * PP) + rowoff + r) * CC + h * DD;
#pragma unroll
    for (int t = 0; t < 4; ++t) attn_out[row + t * 16 + c] = O[t][r] * invl;
  }
}

// ---------------------------------------------------------------------------
// Kernel 3: out = attn_out @ proj_w.T + proj_b   (256 x 768 x 768)
// One wave per 16(M) x 128(N) tile; A fp32 converted inline, B bf16.
// ---------------------------------------------------------------------------
__global__ __launch_bounds__(32) void proj_kernel(
    const float* __restrict__ attn_in, const __bf16* __restrict__ projw_bf,
    const float* __restrict__ proj_b, float* __restrict__ out) {
  const unsigned lane = threadIdx.x & 31u;
  const unsigned nt = blockIdx.x % 6u, mt = blockIdx.x / 6u;
  const unsigned n0 = nt * 128u;
  const unsigned m0 = mt * 16u;
  const unsigned c = lane & 15u;
  const unsigned rowoff = (lane & 16u) ? 8u : 0u;

  v8f acc[8];
#pragma unroll
  for (int t = 0; t < 8; ++t) acc[t] = vzero8();

  const float* abase = attn_in + (size_t)(m0 + c) * CC;
  for (int k0 = 0; k0 < CC; k0 += 32) {
    v16bf a = load_a_f32(abase + k0, lane);
    v16bf bm[8];
#pragma unroll
    for (int t = 0; t < 8; ++t)
      bm[t] = load_b_bf16(projw_bf + (size_t)(n0 + (unsigned)t * 16u + c) * CC + k0, lane);
#pragma unroll
    for (int t = 0; t < 8; ++t) acc[t] = wmma_bf16(a, bm[t], acc[t]);
  }
#pragma unroll
  for (int t = 0; t < 8; ++t) {
    float bias = proj_b[n0 + (unsigned)t * 16u + c];
#pragma unroll
    for (int r = 0; r < 8; ++r)
      out[(size_t)(m0 + rowoff + r) * CC + n0 + (unsigned)t * 16u + c] = acc[t][r] + bias;
  }
}

// ---------------------------------------------------------------------------
extern "C" void kernel_launch(void* const* d_in, const int* in_sizes, int n_in,
                              void* d_out, int out_size, void* d_ws, size_t ws_size,
                              hipStream_t stream) {
  (void)in_sizes; (void)n_in; (void)out_size; (void)ws_size;
  const float* lpn_q  = (const float*)d_in[0];
  const float* x      = (const float*)d_in[1];
  const float* kv_w   = (const float*)d_in[2];
  const float* proj_w = (const float*)d_in[3];
  const float* proj_b = (const float*)d_in[4];
  const float* lsc    = (const float*)d_in[5];

  const size_t kvElems = (size_t)BB * HH * NN * DD;  // 50,331,648
  __bf16* k_bf = (__bf16*)d_ws;
  __bf16* v_t  = k_bf + kvElems;
  float* attn_ws = (float*)(v_t + kvElems);
  __bf16* kvw_bf  = (__bf16*)(attn_ws + (size_t)BB * PP * CC);
  __bf16* projw_bf = kvw_bf + (size_t)2 * CC * CC;

  // one-time weight downconverts
  cvt_f32_bf16_kernel<<<dim3((2 * CC * CC / 8 + 255) / 256), 256, 0, stream>>>(
      kv_w, kvw_bf, 2 * CC * CC / 8);
  cvt_f32_bf16_kernel<<<dim3((CC * CC / 8 + 255) / 256), 256, 0, stream>>>(
      proj_w, projw_bf, CC * CC / 8);

  // KV GEMM: M/128 * Nout/128 = 512 * 12 blocks
  kv_gemm_kernel<<<dim3(512 * 12), 256, 0, stream>>>(x, kvw_bf, k_bf, v_t);
  // Attention: one wave per (b,h)
  attn_kernel<<<dim3(BB * HH), 32, 0, stream>>>(lpn_q, k_bf, v_t, lsc, attn_ws);
  // Projection: 16 M-tiles * 6 N-tiles
  proj_kernel<<<dim3(16 * 6), 32, 0, stream>>>(attn_ws, proj_w ? projw_bf : projw_bf, proj_b, (float*)d_out);
}
